// HQQLinear_39943195853112
// MI455X (gfx1250) — matrix operand order repaired and checked by reading, not compiled, not run on hardware
//
#include <hip/hip_runtime.h>

typedef __attribute__((ext_vector_type(16))) __bf16 v16bf;
typedef __attribute__((ext_vector_type(8)))  float  v8f;
typedef unsigned short u16;

#define IN_F   4096
#define OUT_F  11008
#define NGRP   704512     // IN_F*OUT_F/64
#define M_TOT  8192
#define BM     128
#define BN     128
#define BK     64
#define STR    72         // BK + 8 halves padding (144 B rows)

struct U8 { uint4 a, b; };   // 32 bytes == one v16bf fragment

// round-to-nearest-even f32 -> bf16 packed pair
__device__ __forceinline__ unsigned pack2_bf16(float a, float b) {
  unsigned ua = __builtin_bit_cast(unsigned, a);
  unsigned ub = __builtin_bit_cast(unsigned, b);
  ua = (ua + 0x7FFFu + ((ua >> 16) & 1u)) >> 16;
  ub = (ub + 0x7FFFu + ((ub >> 16) & 1u)) & 0xFFFF0000u;
  return (ua & 0xFFFFu) | ub;
}

__device__ __forceinline__ void wait_asynccnt0() {
#if defined(__has_builtin) && __has_builtin(__builtin_amdgcn_s_wait_asynccnt)
  __builtin_amdgcn_s_wait_asynccnt(0);
#else
  asm volatile("s_wait_asynccnt 0x0" ::: "memory");
#endif
}

// one lane copies 16 B global -> LDS, ASYNCcnt-tracked
__device__ __forceinline__ void async_b128(const u16* gsrc, u16* ldst) {
  unsigned laddr = (unsigned)(unsigned long long)(void*)ldst;  // LDS byte addr = low 32 bits
  asm volatile("global_load_async_to_lds_b128 %0, %1, off"
               :: "v"(laddr), "v"(gsrc) : "memory");
}

// stage a 128-row x 64-col bf16 tile (row stride gstride elems) into padded LDS
__device__ __forceinline__ void issue_tile(const u16* __restrict__ g, long gstride,
                                           u16* lds, int t) {
  #pragma unroll
  for (int j = 0; j < 4; ++j) {
    const int s   = t + j * 256;     // 1024 x 16B segments
    const int row = s >> 3;
    const int seg = (s & 7) * 8;     // elem offset
    async_b128(g + (long)row * gstride + seg, lds + row * STR + seg);
  }
}

// ============ pass 1: dequantize W -> bf16 [OUT_F][IN_F] ============
__global__ __launch_bounds__(256)
void dequant_w_kernel(const int* __restrict__ Wq, const float* __restrict__ scale,
                      const float* __restrict__ zero, u16* __restrict__ Wb) {
  const int o     = blockIdx.x;                     // one output row per block
  const int r     = o / 172;
  const int o0    = o - r * 172;
  const int shift = (r < 32) ? 4 : 0;
  const long qb   = (long)(r & 31) * NGRP + (long)o0 * IN_F;
  const long sb   = (long)o0 * IN_F;
  const int  i0   = threadIdx.x * 16;

  const int*   qp = Wq + qb + i0;
  const float* sp = scale + sb + i0;
  const float* zp = zero + sb + i0;
  u16* dst = Wb + (long)o * IN_F + i0;

  #pragma unroll
  for (int c = 0; c < 4; ++c) {
    int4   q = ((const int4*)qp)[c];
    float4 s = ((const float4*)sp)[c];
    float4 z = ((const float4*)zp)[c];
    float w0 = ((float)((q.x >> shift) & 0xF) - z.x) * s.x;
    float w1 = ((float)((q.y >> shift) & 0xF) - z.y) * s.y;
    float w2 = ((float)((q.z >> shift) & 0xF) - z.z) * s.z;
    float w3 = ((float)((q.w >> shift) & 0xF) - z.w) * s.w;
    *(uint2*)(dst + c * 4) = make_uint2(pack2_bf16(w0, w1), pack2_bf16(w2, w3));
  }
}

// ============ pass 2: x f32 -> bf16 ============
__global__ __launch_bounds__(256)
void cvt_x_kernel(const float* __restrict__ x, u16* __restrict__ xb) {
  const long base = ((long)blockIdx.x * 256 + threadIdx.x) * 16;
  const float* xp = x + base;
  u16* dst = xb + base;
  #pragma unroll
  for (int c = 0; c < 4; ++c) {
    float4 f = ((const float4*)xp)[c];
    *(uint2*)(dst + c * 4) = make_uint2(pack2_bf16(f.x, f.y), pack2_bf16(f.z, f.w));
  }
}

// ============ pass 3: bf16 WMMA GEMM with async double-buffered LDS ============
__global__ __launch_bounds__(256, 2)
void gemm_bf16_async_kernel(const u16* __restrict__ A,   // [8192][4096] bf16
                            const u16* __restrict__ B,   // [11008][4096] bf16
                            const float* __restrict__ bias,
                            float* __restrict__ out) {
  __shared__ __align__(16) u16 ldsA[2][BM * STR];
  __shared__ __align__(16) u16 ldsB[2][BN * STR];

  const int NB_N  = OUT_F / BN;                 // 86
  const int gid   = blockIdx.x;
  const int super = gid / (NB_N * 8);
  const int rem   = gid % (NB_N * 8);
  const int bn    = rem >> 3;
  const int bm    = super * 8 + (rem & 7);      // 8-M supertiles, mloc fastest -> L2 reuse

  const int t     = threadIdx.x;
  const int lane  = t & 31;
  const int wid   = t >> 5;
  const int waveM = wid >> 2;                   // 0..1
  const int waveN = wid & 3;                    // 0..3
  const int l16   = lane & 15;
  const int hi    = lane >> 4;

  const u16* Ab = A + (long)(bm * BM) * IN_F;
  const u16* Bb = B + (long)(bn * BN) * IN_F;

  v8f acc[4][2];
  #pragma unroll
  for (int mt = 0; mt < 4; ++mt)
    #pragma unroll
    for (int nt = 0; nt < 2; ++nt)
      acc[mt][nt] = (v8f){0.f,0.f,0.f,0.f,0.f,0.f,0.f,0.f};

  // prologue: fill buffer 0
  issue_tile(Ab, IN_F, ldsA[0], t);
  issue_tile(Bb, IN_F, ldsB[0], t);

  for (int kb = 0; kb < IN_F; kb += BK) {
    const int cur = (kb >> 6) & 1;

    wait_asynccnt0();          // my async copies into buf[cur] done
    __syncthreads();           // everyone's copies visible; prev compute on buf[cur^1] done

    if (kb + BK < IN_F) {      // overlap next tile copy with compute
      issue_tile(Ab + kb + BK, IN_F, ldsA[cur ^ 1], t);
      issue_tile(Bb + kb + BK, IN_F, ldsB[cur ^ 1], t);
    }

    #pragma unroll
    for (int ks = 0; ks < 2; ++ks) {
      const int k0 = ks * 32;

      v16bf afrag[4];
      #pragma unroll
      for (int mt = 0; mt < 4; ++mt) {
        // A 16x32 bf16, contiguous k order (same as B): low lanes K0-15, high lanes K16-31
        const int e0 = (waveM * 64 + mt * 16 + l16) * STR + k0 + hi * 16;
        U8 u;
        u.a = *(const uint4*)&ldsA[cur][e0];
        u.b = *(const uint4*)&ldsA[cur][e0 + 8];
        afrag[mt] = __builtin_bit_cast(v16bf, u);
      }

      #pragma unroll
      for (int nt = 0; nt < 2; ++nt) {
        // B 32x16: lane = column; contiguous k order (same as A): low lanes K0-15, high lanes K16-31
        const int e0 = (waveN * 32 + nt * 16 + l16) * STR + k0 + hi * 16;
        U8 u;
        u.a = *(const uint4*)&ldsB[cur][e0];
        u.b = *(const uint4*)&ldsB[cur][e0 + 8];
        const v16bf bfrag = __builtin_bit_cast(v16bf, u);
        #pragma unroll
        for (int mt = 0; mt < 4; ++mt)
          acc[mt][nt] = __builtin_amdgcn_wmma_f32_16x16x32_bf16(
              false, afrag[mt], false, bfrag, (short)0, acc[mt][nt], false, false);
      }
    }
  }

  const int rowBase = bm * BM + waveM * 64;
  const int colBase = bn * BN + waveN * 32;
  #pragma unroll
  for (int nt = 0; nt < 2; ++nt) {
    const int col = colBase + nt * 16 + l16;
    const float bv = bias[col];
    #pragma unroll
    for (int mt = 0; mt < 4; ++mt) {
      const int row0 = rowBase + mt * 16 + hi * 8;
      #pragma unroll
      for (int j = 0; j < 8; ++j)
        out[(long)(row0 + j) * OUT_F + col] = acc[mt][nt][j] + bv;
    }
  }
}

// ============ fallback: fused dequant GEMM (round-1 kernel) ============
__global__ __launch_bounds__(256, 2)
void hqq_fused_kernel(const float* __restrict__ x, const int* __restrict__ Wq,
                      const float* __restrict__ scale, const float* __restrict__ zero,
                      const float* __restrict__ bias, float* __restrict__ out) {
  __shared__ __align__(16) u16 ldsA[BM * STR];
  __shared__ __align__(16) u16 ldsB[BN * STR];

  const int NB_N  = OUT_F / BN;
  const int gid   = blockIdx.x;
  const int super = gid / (NB_N * 8);
  const int rem   = gid % (NB_N * 8);
  const int bn    = rem >> 3;
  const int bm    = super * 8 + (rem & 7);

  const int t     = threadIdx.x;
  const int lane  = t & 31;
  const int wid   = t >> 5;
  const int waveM = wid >> 2;
  const int waveN = wid & 3;
  const int l16   = lane & 15;
  const int hi    = lane >> 4;

  const int srow = t >> 1;
  const int skh  = (t & 1) * 32;

  const long xbase = (long)(bm * BM + srow) * IN_F + skh;
  const int  o     = bn * BN + srow;
  const int  r     = o / 172;
  const int  o0    = o - r * 172;
  const int  shift = (r < 32) ? 4 : 0;
  const long wqb   = (long)(r & 31) * NGRP + (long)o0 * IN_F + skh;
  const long szb   = (long)o0 * IN_F + skh;

  v8f acc[4][2];
  #pragma unroll
  for (int mt = 0; mt < 4; ++mt)
    #pragma unroll
    for (int nt = 0; nt < 2; ++nt)
      acc[mt][nt] = (v8f){0.f,0.f,0.f,0.f,0.f,0.f,0.f,0.f};

  for (int kb = 0; kb < IN_F; kb += BK) {
    __syncthreads();
    {
      const float* xp = x + xbase + kb;
      u16* dst = &ldsA[srow * STR + skh];
      #pragma unroll
      for (int c = 0; c < 8; ++c) {
        float4 f = ((const float4*)xp)[c];
        *(uint2*)(dst + c * 4) = make_uint2(pack2_bf16(f.x, f.y), pack2_bf16(f.z, f.w));
      }
    }
    {
      const int*   qp = Wq + wqb + kb;
      const float* sp = scale + szb + kb;
      const float* zp = zero + szb + kb;
      u16* dst = &ldsB[srow * STR + skh];
      #pragma unroll
      for (int c = 0; c < 8; ++c) {
        int4   q = ((const int4*)qp)[c];
        float4 s = ((const float4*)sp)[c];
        float4 z = ((const float4*)zp)[c];
        float w0 = ((float)((q.x >> shift) & 0xF) - z.x) * s.x;
        float w1 = ((float)((q.y >> shift) & 0xF) - z.y) * s.y;
        float w2 = ((float)((q.z >> shift) & 0xF) - z.z) * s.z;
        float w3 = ((float)((q.w >> shift) & 0xF) - z.w) * s.w;
        *(uint2*)(dst + c * 4) = make_uint2(pack2_bf16(w0, w1), pack2_bf16(w2, w3));
      }
    }
    __syncthreads();

    #pragma unroll
    for (int ks = 0; ks < 2; ++ks) {
      const int k0 = ks * 32;
      v16bf afrag[4];
      #pragma unroll
      for (int mt = 0; mt < 4; ++mt) {
        const int e0 = (waveM * 64 + mt * 16 + l16) * STR + k0 + hi * 16;
        U8 u; u.a = *(const uint4*)&ldsA[e0]; u.b = *(const uint4*)&ldsA[e0 + 8];
        afrag[mt] = __builtin_bit_cast(v16bf, u);
      }
      #pragma unroll
      for (int nt = 0; nt < 2; ++nt) {
        const int e0 = (waveN * 32 + nt * 16 + l16) * STR + k0 + hi * 16;
        U8 u; u.a = *(const uint4*)&ldsB[e0]; u.b = *(const uint4*)&ldsB[e0 + 8];
        const v16bf bfrag = __builtin_bit_cast(v16bf, u);
        #pragma unroll
        for (int mt = 0; mt < 4; ++mt)
          acc[mt][nt] = __builtin_amdgcn_wmma_f32_16x16x32_bf16(
              false, afrag[mt], false, bfrag, (short)0, acc[mt][nt], false, false);
      }
    }
  }

  const int rowBase = bm * BM + waveM * 64;
  const int colBase = bn * BN + waveN * 32;
  #pragma unroll
  for (int nt = 0; nt < 2; ++nt) {
    const int col = colBase + nt * 16 + l16;
    const float bv = bias[col];
    #pragma unroll
    for (int mt = 0; mt < 4; ++mt) {
      const int row0 = rowBase + mt * 16 + hi * 8;
      #pragma unroll
      for (int j = 0; j < 8; ++j)
        out[(long)(row0 + j) * OUT_F + col] = acc[mt][nt][j] + bv;
    }
  }
}

extern "C" void kernel_launch(void* const* d_in, const int* in_sizes, int n_in,
                              void* d_out, int out_size, void* d_ws, size_t ws_size,
                              hipStream_t stream) {
  (void)in_sizes; (void)n_in; (void)out_size;
  const float* x     = (const float*)d_in[0];
  const int*   Wq    = (const int*)  d_in[1];
  const float* scale = (const float*)d_in[2];
  const float* zero  = (const float*)d_in[3];
  const float* bias  = (const float*)d_in[4];
  float* out = (float*)d_out;

  const int grid = (M_TOT / BM) * (OUT_F / BN);           // 5504
  const size_t wbytes = (size_t)OUT_F * IN_F * 2;         // 90.2 MB (16B-aligned)
  const size_t xbytes = (size_t)M_TOT * IN_F * 2;         // 67.1 MB

  if (ws_size >= wbytes + xbytes) {
    u16* Wb = (u16*)d_ws;
    u16* xb = (u16*)((char*)d_ws + wbytes);
    dequant_w_kernel<<<OUT_F, 256, 0, stream>>>(Wq, scale, zero, Wb);
    cvt_x_kernel<<<(int)((size_t)M_TOT * IN_F / 4096), 256, 0, stream>>>(x, xb);
    gemm_bf16_async_kernel<<<grid, 256, 0, stream>>>(xb, Wb, bias, out);
  } else {
    hqq_fused_kernel<<<grid, 256, 0, stream>>>(x, Wq, scale, zero, bias, out);
  }
}
